// QuantumAutoEncoder_45853070852455
// MI455X (gfx1250) — matrix operand order repaired and checked by reading, not compiled, not run on hardware
//
#include <hip/hip_runtime.h>

typedef __bf16 bf16;
typedef __attribute__((ext_vector_type(16))) __bf16 v16bf;
typedef __attribute__((ext_vector_type(8)))  __bf16 v8bf;
typedef __attribute__((ext_vector_type(8)))  float  v8f;

// ---------------------------------------------------------------------------
// fp32 -> bf16 elementwise conversion (grid-stride) — for activations (A side)
// ---------------------------------------------------------------------------
__global__ void cvt_f32_bf16(const float* __restrict__ src,
                             bf16* __restrict__ dst, int n) {
  int i = blockIdx.x * blockDim.x + threadIdx.x;
  int stride = gridDim.x * blockDim.x;
  for (; i < n; i += stride) dst[i] = (bf16)src[i];
}

// ---------------------------------------------------------------------------
// fp32 (K,N) -> bf16 transposed (N,K). LDS-tiled so both global streams are
// coalesced. Hoists the B-operand transpose out of the GEMM hot loop: WMMA's
// B fragment wants K-contiguous per column, so we store weights as W^T once.
// ---------------------------------------------------------------------------
__global__ void __launch_bounds__(256)
cvt_transpose_f32_bf16(const float* __restrict__ src, bf16* __restrict__ dst,
                       int K, int N) {
  __shared__ float tile[32][33];
  const int kb = blockIdx.y * 32;
  const int nb = blockIdx.x * 32;
  const int tx = threadIdx.x;  // 0..31
  for (int i = threadIdx.y; i < 32; i += 8)
    tile[i][tx] = src[(size_t)(kb + i) * N + (nb + tx)];
  __syncthreads();
  for (int i = threadIdx.y; i < 32; i += 8)
    dst[(size_t)(nb + i) * K + (kb + tx)] = (bf16)tile[tx][i];
}

// ---------------------------------------------------------------------------
// Tiled bf16 GEMM, f32 accumulate, V_WMMA_F32_16X16X32_BF16.
//   C[M,N] = act(A[M,K] @ B[K,N] + bias[N]),  B supplied pre-transposed (N,K).
//
// Block = 256 threads (8 wave32 waves) -> 128x256 C tile.
// Wave grid 2x4, each wave owns 64x64 = 4x4 WMMA tiles (16 v8f accumulators).
// K staged 32-wide through LDS; both A (M-row,K) and Bt (N-row,K) tiles are
// K-contiguous, so staging is pure global_load_b128 -> ds_store_b128 and the
// fragments are contiguous 16B ds_load_b128 pairs.
//
// Register-staged pipeline: store regs->LDS, barrier, issue next tile's global
// loads, then 16 WMMAs cover the load latency (loadcnt waited at next store).
//
// CDNA5 16-bit A 16x32 lane layout: per lane (row = lane&15),
//   elems 0..7  = K [ 8*half,  8*half+8)
//   elems 8..15 = K [16+8*half, 24+8*half)
// B 32x16 lane layout: per lane (col = lane&15), same K order as A:
// elems 0..7 = K [8*half, 8*half+8), elems 8..15 = K [16+8*half, 24+8*half).
// Requires M%128==0, N%256==0, K%32==0 (holds for all layers here).
// ---------------------------------------------------------------------------
template <typename OutT, bool RELU>
__global__ void __launch_bounds__(256)
gemm_bf16_wmma(const bf16* __restrict__ A, const bf16* __restrict__ Bt,
               const float* __restrict__ bias, OutT* __restrict__ C,
               int M, int N, int K) {
  __shared__ __align__(16) bf16 As[128][40];  // +8 pad keeps 16B align, skews banks
  __shared__ __align__(16) bf16 Bs[256][40];  // Bs[n][k]

  const int tid  = threadIdx.x;
  const int lane = tid & 31;
  const int wave = tid >> 5;
  const int wr   = wave >> 2;   // 0..1 -> 64 rows each
  const int wc   = wave & 3;    // 0..3 -> 64 cols each
  const int half = lane >> 4;   // 0..1
  const int nl   = lane & 15;

  const int row0 = blockIdx.y * 128;
  const int col0 = blockIdx.x * 256;

  // staging split: A 128x32 -> 16 elems/thread; Bt 256x32 -> 32 elems/thread
  const int a_row = tid >> 1;
  const int a_chk = (tid & 1) * 16;
  const int b_row = tid;

  const bf16* gA = A  + (size_t)(row0 + a_row) * K + a_chk;
  const bf16* gB = Bt + (size_t)(col0 + b_row) * K;

  v8f acc[4][4] = {};

  const int ksteps = K >> 5;

  // ---- pipeline prologue: load K-step 0 into stage regs ----
  v8bf sa0 = *(const v8bf*)(gA);
  v8bf sa1 = *(const v8bf*)(gA + 8);
  v8bf sb0 = *(const v8bf*)(gB);
  v8bf sb1 = *(const v8bf*)(gB + 8);
  v8bf sb2 = *(const v8bf*)(gB + 16);
  v8bf sb3 = *(const v8bf*)(gB + 24);

  for (int ks = 0; ks < ksteps; ++ks) {
    // ---- stage regs -> LDS ----
    *(v8bf*)&As[a_row][a_chk]     = sa0;
    *(v8bf*)&As[a_row][a_chk + 8] = sa1;
    *(v8bf*)&Bs[b_row][0]  = sb0;
    *(v8bf*)&Bs[b_row][8]  = sb1;
    *(v8bf*)&Bs[b_row][16] = sb2;
    *(v8bf*)&Bs[b_row][24] = sb3;
    __syncthreads();

    // ---- issue next tile's global loads (overlaps with WMMA below) ----
    if (ks + 1 < ksteps) {
      const int k1 = (ks + 1) << 5;
      sa0 = *(const v8bf*)(gA + k1);
      sa1 = *(const v8bf*)(gA + k1 + 8);
      sb0 = *(const v8bf*)(gB + k1);
      sb1 = *(const v8bf*)(gB + k1 + 8);
      sb2 = *(const v8bf*)(gB + k1 + 16);
      sb3 = *(const v8bf*)(gB + k1 + 24);
      if (ks + 2 < ksteps) {
        __builtin_prefetch(gA + k1 + 32, 0, 2);  // global_prefetch_b8
        __builtin_prefetch(gB + k1 + 32, 0, 2);
      }
    }

    // ---- B fragments for this wave's 4 column tiles ----
    v16bf bfrag[4];
#pragma unroll
    for (int j = 0; j < 4; ++j) {
      const bf16* p = &Bs[wc * 64 + j * 16 + nl][8 * half];
      ((v8bf*)&bfrag[j])[0] = *(const v8bf*)p;
      ((v8bf*)&bfrag[j])[1] = *(const v8bf*)(p + 16);
    }
    // ---- A fragments + 16 WMMAs ----
#pragma unroll
    for (int i = 0; i < 4; ++i) {
      const bf16* pa = &As[wr * 64 + i * 16 + nl][0];
      v16bf afrag;
      ((v8bf*)&afrag)[0] = *(const v8bf*)(pa + 8 * half);
      ((v8bf*)&afrag)[1] = *(const v8bf*)(pa + 16 + 8 * half);
#pragma unroll
      for (int j = 0; j < 4; ++j) {
        acc[i][j] = __builtin_amdgcn_wmma_f32_16x16x32_bf16(
            false, afrag, false, bfrag[j], (short)0, acc[i][j], false, false);
      }
    }
    __syncthreads();
  }

  // ---- epilogue: bias (+ReLU) + store ----
  // C/D layout: VGPR r -> row (half*8 + r), lane column = nl.
#pragma unroll
  for (int i = 0; i < 4; ++i) {
#pragma unroll
    for (int j = 0; j < 4; ++j) {
      const int col = col0 + wc * 64 + j * 16 + nl;
      const float bs = bias[col];
#pragma unroll
      for (int r = 0; r < 8; ++r) {
        const int m = row0 + wr * 64 + i * 16 + half * 8 + r;
        float v = acc[i][j][r] + bs;
        if (RELU) v = v > 0.0f ? v : 0.0f;
        C[(size_t)m * N + col] = (OutT)v;
      }
    }
  }
}

// ---------------------------------------------------------------------------
// Butterfly of Givens rotations over last dim (n=256, 8 stages).
// One 128-thread block per row; one pair per thread per stage.
// Stage s: pair p -> indices (blk*2*stride + j, +stride), theta idx = s*128+p.
// ---------------------------------------------------------------------------
__global__ void __launch_bounds__(128)
butterfly_kernel(const bf16* __restrict__ in, const float* __restrict__ theta,
                 bf16* __restrict__ out) {
  __shared__ float row[256];
  const int r = blockIdx.x;
  const int t = threadIdx.x;  // 0..127

  row[t]       = (float)in[(size_t)r * 256 + t];
  row[t + 128] = (float)in[(size_t)r * 256 + t + 128];
  __syncthreads();

#pragma unroll
  for (int s = 0; s < 8; ++s) {
    const int stride = 1 << s;
    const int blk = t >> s;
    const int j   = t & (stride - 1);
    const int i0  = blk * 2 * stride + j;
    const int i1  = i0 + stride;
    const float th = theta[s * 128 + t];
    const float c  = __builtin_cosf(th);
    const float sn = __builtin_sinf(th);
    const float a = row[i0];
    const float b = row[i1];
    // each element belongs to exactly one pair per stage -> no intra-stage hazard
    row[i0] = c * a - sn * b;
    row[i1] = sn * a + c * b;
    __syncthreads();
  }

  out[(size_t)r * 256 + t]       = (bf16)row[t];
  out[(size_t)r * 256 + t + 128] = (bf16)row[t + 128];
}

// ---------------------------------------------------------------------------
// Launcher
// ---------------------------------------------------------------------------
extern "C" void kernel_launch(void* const* d_in, const int* in_sizes, int n_in,
                              void* d_out, int out_size, void* d_ws, size_t ws_size,
                              hipStream_t stream) {
  (void)in_sizes; (void)n_in; (void)out_size; (void)ws_size;

  const float* x     = (const float*)d_in[0];
  const float* w0    = (const float*)d_in[1];
  const float* b0    = (const float*)d_in[2];
  const float* w1    = (const float*)d_in[3];
  const float* b1    = (const float*)d_in[4];
  const float* theta = (const float*)d_in[5];
  const float* dw    = (const float*)d_in[6];
  const float* db    = (const float*)d_in[7];
  const float* ow    = (const float*)d_in[8];
  const float* ob    = (const float*)d_in[9];

  const int Bm = 16384;
  const int D0 = 4096, D1 = 1024, D2 = 256;

  // workspace carve-out (bf16 copies + intermediates) ~236 MB
  char* ws = (char*)d_ws;
  size_t off = 0;
  auto carve = [&](size_t bytes) -> void* {
    void* p = ws + off;
    off = (off + bytes + 255) & ~(size_t)255;
    return p;
  };
  bf16* Xb  = (bf16*)carve((size_t)Bm * D0 * 2);
  bf16* W0t = (bf16*)carve((size_t)D0 * D1 * 2);  // (D1, D0)
  bf16* W1t = (bf16*)carve((size_t)D1 * D2 * 2);  // (D2, D1)
  bf16* DWt = (bf16*)carve((size_t)D2 * D1 * 2);  // (D1, D2)
  bf16* OWt = (bf16*)carve((size_t)D1 * D0 * 2);  // (D0, D1)
  bf16* H1  = (bf16*)carve((size_t)Bm * D1 * 2);
  bf16* H2  = (bf16*)carve((size_t)Bm * D2 * 2);
  bf16* Bf  = (bf16*)carve((size_t)Bm * D2 * 2);
  bf16* H3  = (bf16*)carve((size_t)Bm * D1 * 2);

  // activations: elementwise convert; weights: convert + transpose to (N,K)
  cvt_f32_bf16<<<2048, 256, 0, stream>>>(x, Xb, Bm * D0);
  {
    dim3 tb(32, 8);
    cvt_transpose_f32_bf16<<<dim3(D1 / 32, D0 / 32), tb, 0, stream>>>(w0, W0t, D0, D1);
    cvt_transpose_f32_bf16<<<dim3(D2 / 32, D1 / 32), tb, 0, stream>>>(w1, W1t, D1, D2);
    cvt_transpose_f32_bf16<<<dim3(D1 / 32, D2 / 32), tb, 0, stream>>>(dw, DWt, D2, D1);
    cvt_transpose_f32_bf16<<<dim3(D0 / 32, D1 / 32), tb, 0, stream>>>(ow, OWt, D1, D0);
  }

  const dim3 blk(256);
  // h1 = relu(x @ enc_w0 + b0)
  gemm_bf16_wmma<bf16, true><<<dim3(D1 / 256, Bm / 128), blk, 0, stream>>>(
      Xb, W0t, b0, H1, Bm, D1, D0);
  // h2 = relu(h1 @ enc_w1 + b1)
  gemm_bf16_wmma<bf16, true><<<dim3(D2 / 256, Bm / 128), blk, 0, stream>>>(
      H1, W1t, b1, H2, Bm, D2, D1);
  // butterfly rotation over last dim 256
  butterfly_kernel<<<Bm, 128, 0, stream>>>(H2, theta, Bf);
  // h3 = relu(butterfly @ dec_w0 + db)
  gemm_bf16_wmma<bf16, true><<<dim3(D1 / 256, Bm / 128), blk, 0, stream>>>(
      Bf, DWt, db, H3, Bm, D1, D2);
  // out = h3 @ out_w + ob   (fp32 output)
  gemm_bf16_wmma<float, false><<<dim3(D0 / 256, Bm / 128), blk, 0, stream>>>(
      H3, OWt, ob, (float*)d_out, Bm, D0, D1);
}
